// SPCC_90288802497140
// MI455X (gfx1250) — compile-verified, matrix operand order checked by reading, not run on hardware
//
#include <hip/hip_runtime.h>
#include <hip/hip_bf16.h>

typedef __attribute__((ext_vector_type(16))) _Float16 v16h;
typedef __attribute__((ext_vector_type(8)))  float    v8f;

#define U_NUM   16384
#define I_NUM   8192
#define BATCH   4096
#define PD      64
#define SD      64
#define USEG    8
#define UCHUNKS 512   // U_NUM / 32
#define ITILES  512   // I_NUM / 16

// K-index held by vector element e for a lane-half, per CDNA5 16-bit A/B WMMA layout
__device__ __forceinline__ int kmap(int e, int half) {
  return (e < 8 ? e : e + 8) + half * 8;
}

__device__ __forceinline__ v8f wmma_f16(v16h a, v16h b, v8f c) {
  return __builtin_amdgcn_wmma_f32_16x16x32_f16(false, a, false, b, (short)0, c,
                                                false, false);
}

// fast sigmoid: v_exp_f32 + v_rcp_f32 (TRANS ops, co-execute with WMMA)
__device__ __forceinline__ float fast_sigmoid(float x) {
  return __builtin_amdgcn_rcpf(1.0f + __expf(-x));
}

// ---------------- keys = Suser @ att_W^T + att_b  (tiny, VALU) ----------------
__global__ void keys_kernel(const float* __restrict__ Suser,
                            const float* __restrict__ attW,
                            const float* __restrict__ attb,
                            float* __restrict__ keys) {
  int tid = blockIdx.x * blockDim.x + threadIdx.x;  // U_NUM*64 threads
  int u = tid >> 6, k = tid & 63;
  const float* xr = Suser + u * SD;
  const float* wr = attW + k * SD;
  float acc = attb[k];
#pragma unroll
  for (int j = 0; j < SD; ++j) acc += xr[j] * wr[j];
  keys[u * SD + k] = acc;
}

// ---------------- pack kernels: f32 row-major -> f16 WMMA lane layouts ----------------
// keysA[uc][b][h][lane]: A-matrix of GEMM1^T (rows = users of chunk)
__global__ void pack_keysA(const float* __restrict__ keys, v16h* __restrict__ out) {
  int tid = blockIdx.x * blockDim.x + threadIdx.x;  // 512*2*2*32
  int lane = tid & 31, rest = tid >> 5;
  int h = rest & 1, b = (rest >> 1) & 1, uc = rest >> 2;
  int half = lane >> 4;
  const float* src = keys + (uc * 32 + b * 16 + (lane & 15)) * SD + h * 32;
  v16h v;
#pragma unroll
  for (int e = 0; e < 16; ++e) v[e] = (_Float16)src[kmap(e, half)];
  out[tid] = v;
}

// sitemB[it][h][lane]: B-matrix of GEMM1^T (cols = items of tile)
__global__ void pack_sitemB(const float* __restrict__ Sitem, v16h* __restrict__ out) {
  int tid = blockIdx.x * blockDim.x + threadIdx.x;  // 512*2*32
  int lane = tid & 31, rest = tid >> 5;
  int h = rest & 1, it = rest >> 1;
  int half = lane >> 4;
  const float* src = Sitem + (it * 16 + (lane & 15)) * SD + h * 32;
  v16h v;
#pragma unroll
  for (int e = 0; e < 16; ++e) v[e] = (_Float16)src[kmap(e, half)];
  out[tid] = v;
}

// suserAT[uc][mb][lane]: A-matrix of GEMM2^T (rows = embedding dims, K = users)
__global__ void pack_suserAT(const float* __restrict__ Suser, v16h* __restrict__ out) {
  int tid = blockIdx.x * blockDim.x + threadIdx.x;  // 512*4*32
  int lane = tid & 31, rest = tid >> 5;
  int mb = rest & 3, uc = rest >> 2;
  int half = lane >> 4;
  int col = mb * 16 + (lane & 15);
  v16h v;
#pragma unroll
  for (int e = 0; e < 16; ++e)
    v[e] = (_Float16)Suser[(uc * 32 + kmap(e, half)) * SD + col];
  out[tid] = v;
}

// WA[l][mb][h][lane]: MLP weights (L,64,64) as A-matrix (rows = out dims)
__global__ void pack_W(const float* __restrict__ W, v16h* __restrict__ out) {
  int tid = blockIdx.x * blockDim.x + threadIdx.x;  // 2*4*2*32 = 512
  int lane = tid & 31, rest = tid >> 5;
  int h = rest & 1, mb = (rest >> 1) & 3, l = rest >> 3;
  int half = lane >> 4;
  const float* src = W + (l * 64 + mb * 16 + (lane & 15)) * 64 + h * 32;
  v16h v;
#pragma unroll
  for (int e = 0; e < 16; ++e) v[e] = (_Float16)src[kmap(e, half)];
  out[tid] = v;
}

// ---------------- fused relation: qk^T = keys@Sitem^T, rel^T += Suser^T @ (sig(qk)*mask)^T ----------------
__global__ void __launch_bounds__(32)
relation_kernel(const v16h* __restrict__ keysA, const v16h* __restrict__ sitemB,
                const v16h* __restrict__ suserAT, const float* __restrict__ mask,
                float* __restrict__ relP) {
  int it   = blockIdx.x & (ITILES - 1);
  int seg  = blockIdx.x >> 9;
  int lane = threadIdx.x;
  int n16 = lane & 15, half = lane >> 4;

  v16h sB0 = sitemB[it * 64 + lane];        // k-half 0
  v16h sB1 = sitemB[it * 64 + 32 + lane];   // k-half 1
  v8f r0 = {}, r1 = {}, r2 = {}, r3 = {};

  const float* mbase = mask + (size_t)(it * 16 + n16) * U_NUM + half * 8;
  int uc0 = seg * (UCHUNKS / USEG), uc1 = uc0 + (UCHUNKS / USEG);

  for (int uc = uc0; uc < uc1; ++uc) {
    // GEMM1^T: qk^T tiles (users b*16.. x items), K = SD in two halves
    v8f q0 = {}, q1 = {};
    q0 = wmma_f16(keysA[uc * 128 + lane], sB0, q0);
    q0 = wmma_f16(keysA[uc * 128 + 32 + lane], sB1, q0);
    q1 = wmma_f16(keysA[uc * 128 + 64 + lane], sB0, q1);
    q1 = wmma_f16(keysA[uc * 128 + 96 + lane], sB1, q1);

    const float* mrow = mbase + uc * 32;
    __builtin_prefetch(mrow + 32, 0, 1);  // next u-chunk of this mask row
    const float4* mp0 = (const float4*)mrow;         // users +0..7  (tile b=0)
    const float4* mp1 = (const float4*)(mrow + 16);  // users +16..23 (tile b=1)
    float4 a0 = mp0[0], a1 = mp0[1], b0 = mp1[0], b1 = mp1[1];
    float mk0[8] = {a0.x, a0.y, a0.z, a0.w, a1.x, a1.y, a1.z, a1.w};
    float mk1[8] = {b0.x, b0.y, b0.z, b0.w, b1.x, b1.y, b1.z, b1.w};

    // sigmoid * mask, f32 D-layout -> f16 B-layout for GEMM2^T (register-only)
    v16h bB;
#pragma unroll
    for (int r = 0; r < 8; ++r) {
      bB[r]     = (_Float16)(fast_sigmoid(q0[r]) * mk0[r]);
      bB[r + 8] = (_Float16)(fast_sigmoid(q1[r]) * mk1[r]);
    }

    // GEMM2^T: rel^T[dims x items] += Suser^T @ qkm^T, K = 32 users
    r0 = wmma_f16(suserAT[uc * 128 + lane], bB, r0);
    r1 = wmma_f16(suserAT[uc * 128 + 32 + lane], bB, r1);
    r2 = wmma_f16(suserAT[uc * 128 + 64 + lane], bB, r2);
    r3 = wmma_f16(suserAT[uc * 128 + 96 + lane], bB, r3);
  }

  // store rel partial, transposed back: rel[item][dim]
  float* orow = relP + ((size_t)seg * I_NUM + it * 16 + n16) * SD + half * 8;
  v8f acc[4] = {r0, r1, r2, r3};
#pragma unroll
  for (int mb = 0; mb < 4; ++mb) {
    float4 lo = make_float4(acc[mb][0], acc[mb][1], acc[mb][2], acc[mb][3]);
    float4 hi = make_float4(acc[mb][4], acc[mb][5], acc[mb][6], acc[mb][7]);
    *(float4*)(orow + mb * 16)     = lo;
    *(float4*)(orow + mb * 16 + 4) = hi;
  }
}

__global__ void reduce_rel(const float* __restrict__ relP, float* __restrict__ rel) {
  int tid = blockIdx.x * blockDim.x + threadIdx.x;  // I_NUM*SD
  float s = 0.f;
#pragma unroll
  for (int g = 0; g < USEG; ++g) s += relP[(size_t)g * I_NUM * SD + tid];
  rel[tid] = s;
}

// ---------------- dense 2-layer MLP over a table, transposed-chained WMMA ----------------
__global__ void __launch_bounds__(32)
mlp_kernel(const float* __restrict__ in, const v16h* __restrict__ WA,
           const float* __restrict__ bias, float* __restrict__ out) {
  int t0 = blockIdx.x * 16;
  int lane = threadIdx.x, n16 = lane & 15, half = lane >> 4;

  // B-operand of layer1: x0^T (cols = rows of the table tile)
  const float* xr = in + (t0 + n16) * PD;
  v16h xB0, xB1;
#pragma unroll
  for (int e = 0; e < 16; ++e) {
    int k = kmap(e, half);
    xB0[e] = (_Float16)xr[k];
    xB1[e] = (_Float16)xr[k + 32];
  }

  // layer 1: x1^T = W0 @ x0^T + b0
  v8f d[4];
#pragma unroll
  for (int mb = 0; mb < 4; ++mb) {
    v8f z = {};
    z = wmma_f16(WA[(mb * 2 + 0) * 32 + lane], xB0, z);
    z = wmma_f16(WA[(mb * 2 + 1) * 32 + lane], xB1, z);
#pragma unroll
    for (int r = 0; r < 8; ++r) z[r] += bias[mb * 16 + half * 8 + r];
    d[mb] = z;
  }

  // D-layout of layer1 is exactly the B-layout of layer2 (register re-pack only)
  v16h yB0, yB1;
#pragma unroll
  for (int e = 0; e < 8; ++e) {
    yB0[e]     = (_Float16)d[0][e];
    yB0[e + 8] = (_Float16)d[1][e];
    yB1[e]     = (_Float16)d[2][e];
    yB1[e + 8] = (_Float16)d[3][e];
  }

  const float* b1 = bias + PD;
  float* orow = out + (t0 + n16) * PD + half * 8;
#pragma unroll
  for (int mb = 0; mb < 4; ++mb) {
    v8f z = {};
    z = wmma_f16(WA[((4 + mb) * 2 + 0) * 32 + lane], yB0, z);
    z = wmma_f16(WA[((4 + mb) * 2 + 1) * 32 + lane], yB1, z);
#pragma unroll
    for (int r = 0; r < 8; ++r) z[r] += b1[mb * 16 + half * 8 + r];
    float4 lo = make_float4(z[0], z[1], z[2], z[3]);
    float4 hi = make_float4(z[4], z[5], z[6], z[7]);
    *(float4*)(orow + mb * 16)     = lo;
    *(float4*)(orow + mb * 16 + 4) = hi;
  }
}

// ---------------- final gather + dot-product scoring ----------------
__global__ void score_kernel(const int* __restrict__ user, const int* __restrict__ pit,
                             const int* __restrict__ nit,
                             const float* __restrict__ puT, const float* __restrict__ piT,
                             const float* __restrict__ Suser, const float* __restrict__ Sitem,
                             const float* __restrict__ rel, float* __restrict__ out) {
  int b = blockIdx.x * blockDim.x + threadIdx.x;
  if (b >= BATCH) return;
  int u = user[b], p = pit[b], n = nit[b];
  const float* pu = puT + u * PD;
  const float* pp = piT + p * PD;
  const float* pn = piT + n * PD;
  const float* su = Suser + u * SD;
  const float* sp = Sitem + p * SD;
  const float* sn = Sitem + n * SD;
  const float* rp = rel + p * SD;
  const float* rn = rel + n * SD;
  float Pp = 0, Pn = 0, Sp = 0, Sn = 0;
#pragma unroll
  for (int j = 0; j < PD; ++j) {
    Pp += pu[j] * pp[j];
    Pn += pu[j] * pn[j];
    Sp += (su[j] + rp[j]) * sp[j];
    Sn += (su[j] + rn[j]) * sn[j];
  }
  out[b] = Pp;
  out[BATCH + b] = Pn;
  out[2 * BATCH + b] = Sp;
  out[3 * BATCH + b] = Sn;
  out[4 * BATCH + b] = (float)u;
}

extern "C" void kernel_launch(void* const* d_in, const int* in_sizes, int n_in,
                              void* d_out, int out_size, void* d_ws, size_t ws_size,
                              hipStream_t stream) {
  const int*   user      = (const int*)d_in[0];
  const int*   p_item    = (const int*)d_in[1];
  const int*   n_item    = (const int*)d_in[2];
  const float* mask      = (const float*)d_in[3];
  const float* Puser_tab = (const float*)d_in[4];
  const float* Pitem_tab = (const float*)d_in[5];
  const float* Suser_tab = (const float*)d_in[6];
  const float* Sitem_tab = (const float*)d_in[7];
  const float* att_W     = (const float*)d_in[8];
  const float* att_b     = (const float*)d_in[9];
  const float* u_mlp_W   = (const float*)d_in[10];
  const float* u_mlp_b   = (const float*)d_in[11];
  const float* i_mlp_W   = (const float*)d_in[12];
  const float* i_mlp_b   = (const float*)d_in[13];
  float* out = (float*)d_out;

  char* ws = (char*)d_ws;
  float* keys    = (float*)(ws + 0);          //  4 MB
  v16h*  keysA   = (v16h*)(ws + 4194304);     //  2 MB
  v16h*  sitemB  = (v16h*)(ws + 6291456);     //  1 MB
  v16h*  suserAT = (v16h*)(ws + 7340032);     //  2 MB
  v16h*  WuA     = (v16h*)(ws + 9437184);     // 16 KB
  v16h*  WiA     = (v16h*)(ws + 9453568);     // 16 KB
  float* relP    = (float*)(ws + 9469952);    // 16 MB
  float* rel     = (float*)(ws + 26247168);   //  2 MB
  float* puT     = (float*)(ws + 28344320);   //  4 MB
  float* piT     = (float*)(ws + 32538624);   //  2 MB

  keys_kernel<<<(U_NUM * 64) / 256, 256, 0, stream>>>(Suser_tab, att_W, att_b, keys);
  pack_keysA<<<256, 256, 0, stream>>>(keys, keysA);
  pack_sitemB<<<128, 256, 0, stream>>>(Sitem_tab, sitemB);
  pack_suserAT<<<256, 256, 0, stream>>>(Suser_tab, suserAT);
  pack_W<<<2, 256, 0, stream>>>(u_mlp_W, WuA);
  pack_W<<<2, 256, 0, stream>>>(i_mlp_W, WiA);

  relation_kernel<<<ITILES * USEG, 32, 0, stream>>>(keysA, sitemB, suserAT, mask, relP);
  reduce_rel<<<(I_NUM * SD) / 256, 256, 0, stream>>>(relP, rel);

  mlp_kernel<<<U_NUM / 16, 32, 0, stream>>>(Puser_tab, WuA, u_mlp_b, puT);
  mlp_kernel<<<I_NUM / 16, 32, 0, stream>>>(Pitem_tab, WiA, i_mlp_b, piT);

  score_kernel<<<BATCH / 256, 256, 0, stream>>>(user, p_item, n_item, puT, piT,
                                                Suser_tab, Sitem_tab, rel, out);
}
